// Evaluate_62234076119449
// MI455X (gfx1250) — 8-bit matrix operand order repaired and checked by reading, compiled, not run on hardware
//
#include <hip/hip_runtime.h>
#include <hip/hip_bf16.h>
#include <stdint.h>

typedef int v8i __attribute__((ext_vector_type(8)));

#define BS    4
#define P_CH  256
#define GT_CH 64
#define HW    (512*512)
#define NCLS  81

#define PBLK   64
#define NPBLK  (P_CH/PBLK)
#define HCHUNK 256
#define ITERS  8
#define HPERWG (HCHUNK*ITERS)
#define NHC    (HW/HPERWG)

#define PSTRIDE 272
#define TSTRIDE 80

#define WS_INTP   0
#define WS_PSUM   (WS_INTP + BS*P_CH*GT_CH)
#define WS_TSUM   (WS_PSUM + BS*P_CH)
#define WS_SCORE  (WS_TSUM + BS*GT_CH)
#define WS_CLS    (WS_SCORE + BS*P_CH)
#define WS_STATS  (WS_CLS + BS*P_CH)
#define WS_ZERO_N WS_SCORE

__global__ void zero_kernel(int* __restrict__ ws, int n) {
    int i = blockIdx.x * blockDim.x + threadIdx.x;
    if (i < n) ws[i] = 0;
}

__global__ __launch_bounds__(128) void iou_wmma_kernel(const float* __restrict__ pred,
                                                       const float* __restrict__ tgt,
                                                       int* __restrict__ ws) {
    __shared__ unsigned char predLDS[PBLK][PSTRIDE];
    __shared__ unsigned char tgtT[HCHUNK][TSTRIDE];

    const int hc   = blockIdx.x;
    const int pblk = blockIdx.y;
    const int b    = blockIdx.z;
    const int t    = threadIdx.x;
    const int row  = t >> 1;
    const int half = t & 1;
    const int lane = t & 31;
    const int wave = t >> 5;
    const int m    = lane & 15;
    const int hw2  = lane >> 4;
    const int pbase = pblk * PBLK;

    v8i acc[4];
    #pragma unroll
    for (int i = 0; i < 4; ++i) acc[i] = (v8i){0,0,0,0,0,0,0,0};

    int cntP = 0, cntT = 0;

    const size_t hb0 = (size_t)hc * HPERWG + (size_t)half * 128;
    const float* predRowP = pred + (size_t)(b * P_CH + pbase + row) * HW + hb0;
    const float* tgtRowP  = tgt  + (size_t)(b * GT_CH + row)        * HW + hb0;

    for (int it = 0; it < ITERS; ++it) {
        const float4* pr = (const float4*)(predRowP + (size_t)it * HCHUNK);
        const float4* tr = (const float4*)(tgtRowP  + (size_t)it * HCHUNK);
        if (it + 1 < ITERS) {
            __builtin_prefetch((const char*)pr + 1024, 0, 0);
            __builtin_prefetch((const char*)tr + 1024, 0, 0);
        }
        #pragma unroll 4
        for (int j = 0; j < 32; ++j) {
            float4 v = pr[j];
            unsigned int b0 = v.x > 0.5f, b1 = v.y > 0.5f, b2 = v.z > 0.5f, b3 = v.w > 0.5f;
            cntP += (int)(b0 + b1 + b2 + b3);
            *(unsigned int*)&predLDS[row][half * 128 + 4 * j] =
                b0 | (b1 << 8) | (b2 << 16) | (b3 << 24);

            float4 w = tr[j];
            unsigned char c0 = w.x != 0.0f, c1 = w.y != 0.0f, c2 = w.z != 0.0f, c3 = w.w != 0.0f;
            cntT += (int)c0 + (int)c1 + (int)c2 + (int)c3;
            int hl = half * 128 + 4 * j;
            tgtT[hl + 0][row] = c0;
            tgtT[hl + 1][row] = c1;
            tgtT[hl + 2][row] = c2;
            tgtT[hl + 3][row] = c3;
        }
        __syncthreads();

        const int g0 = wave * 16;
        #pragma unroll
        for (int ks = 0; ks < 4; ++ks) {
            const int kb = ks * 64;
            const int K1 = kb + m + 16 * hw2;
            v8i bf;
            {
                (void)K1;
                #pragma unroll
                for (int q = 0; q < 8; ++q) {
                    const int k0 = kb + 8 * hw2 + 4 * (q & 1) + 16 * (q >> 1);
                    bf[q] = (int)((unsigned int)tgtT[k0][g0 + m]
                          | ((unsigned int)tgtT[k0 + 1][g0 + m] << 8)
                          | ((unsigned int)tgtT[k0 + 2][g0 + m] << 16)
                          | ((unsigned int)tgtT[k0 + 3][g0 + m] << 24));
                }
            }
            const int abase = kb + 8 * hw2;
            #pragma unroll
            for (int pa = 0; pa < 4; ++pa) {
                const unsigned char* ar = &predLDS[pa * 16 + m][0];
                unsigned long long x0 = *(const unsigned long long*)(ar + abase);
                unsigned long long x1 = *(const unsigned long long*)(ar + abase + 16);
                unsigned long long x2 = *(const unsigned long long*)(ar + abase + 32);
                unsigned long long x3 = *(const unsigned long long*)(ar + abase + 48);
                v8i af;
                af[0] = (int)x0; af[1] = (int)(x0 >> 32);
                af[2] = (int)x1; af[3] = (int)(x1 >> 32);
                af[4] = (int)x2; af[5] = (int)(x2 >> 32);
                af[6] = (int)x3; af[7] = (int)(x3 >> 32);
                acc[pa] = __builtin_amdgcn_wmma_i32_16x16x64_iu8(
                    false, af, false, bf, acc[pa], false, false);
            }
        }
        __syncthreads();
    }

    int* intp = ws + WS_INTP;
    const int gidx = wave * 16 + m;
    #pragma unroll
    for (int pa = 0; pa < 4; ++pa) {
        #pragma unroll
        for (int r = 0; r < 8; ++r) {
            int pidx = pbase + pa * 16 + r + 8 * hw2;
            atomicAdd(&intp[(b * P_CH + pidx) * GT_CH + gidx], acc[pa][r]);
        }
    }
    atomicAdd(&ws[WS_PSUM + b * P_CH + pbase + row], cntP);
    if (pblk == 0) atomicAdd(&ws[WS_TSUM + b * GT_CH + row], cntT);
}

__global__ void score_kernel(const float* __restrict__ logits, int* __restrict__ ws) {
    const int rowid = blockIdx.x;
    const int lane  = threadIdx.x;
    const float* l = logits + (size_t)rowid * NCLS;

    float bv = -1e30f; int bi = 0;
    for (int c = lane; c < NCLS; c += 32) {
        float x = l[c];
        if (x > bv) { bv = x; bi = c; }
    }
    for (int off = 16; off >= 1; off >>= 1) {
        float ov = __shfl_down(bv, off, 32);
        int   oi = __shfl_down(bi, off, 32);
        if (ov > bv || (ov == bv && oi < bi)) { bv = ov; bi = oi; }
    }
    bv = __shfl(bv, 0, 32);
    bi = __shfl(bi, 0, 32);

    float s = 0.0f;
    for (int c = lane; c < NCLS; c += 32) s += expf(l[c] - bv);
    for (int off = 16; off >= 1; off >>= 1) s += __shfl_down(s, off, 32);

    if (lane == 0) {
        ((float*)ws)[WS_SCORE + rowid] = 1.0f / s;
        ws[WS_CLS + rowid] = bi;
    }
}

__global__ void match_kernel(const int* __restrict__ tcls, int* __restrict__ ws) {
    const int b = blockIdx.x;
    if (threadIdx.x != 0) return;

    const int*   intp  = ws + WS_INTP + b * P_CH * GT_CH;
    const int*   psum  = ws + WS_PSUM + b * P_CH;
    const int*   tsum  = ws + WS_TSUM + b * GT_CH;
    const float* score = (const float*)ws + WS_SCORE + b * P_CH;
    const int*   cls   = ws + WS_CLS + b * P_CH;
    const int*   tc    = tcls + b * GT_CH;

    bool used[P_CH];
    bool claimed[GT_CH];
    for (int p = 0; p < P_CH; ++p) used[p] = false;
    for (int g = 0; g < GT_CH; ++g) claimed[g] = false;

    float tp = 0.0f, fp = 0.0f;
    for (int it = 0; it < P_CH; ++it) {
        float bsv = -1e30f; int pk = 0;
        for (int p = 0; p < P_CH; ++p)
            if (!used[p] && score[p] > bsv) { bsv = score[p]; pk = p; }
        used[pk] = true;

        float ps = (float)psum[pk];
        bool skip = (cls[pk] == 0) || (ps < 1.0f) || (score[pk] < 0.5f);

        float rb = -1.0f; int gk = 0;
        for (int g = 0; g < GT_CH; ++g) {
            float ii = (float)intp[pk * GT_CH + g];
            float uu = ps + (float)tsum[g] - ii;
            float iou = claimed[g] ? 0.0f : ii / (uu + 0.01f);
            if (iou > rb) { rb = iou; gk = g; }
        }
        bool hit = (rb >= 0.5f) && (cls[pk] == tc[gk]) && !skip;
        tp += hit ? 1.0f : 0.0f;
        fp += (!hit && !skip) ? 1.0f : 0.0f;
        if (hit) claimed[gk] = true;
    }

    float tt = 0.0f;
    for (int g = 0; g < GT_CH; ++g) tt += (tc[g] > 0) ? 1.0f : 0.0f;

    float* st = (float*)ws + WS_STATS + b * 3;
    st[0] = tp; st[1] = fp; st[2] = tt;
}

__global__ void final_kernel(const int* __restrict__ ws, float* __restrict__ out) {
    const float* st = (const float*)ws + WS_STATS;
    float tp = 0.0f, fp = 0.0f, tt = 0.0f;
    for (int b = 0; b < BS; ++b) { tp += st[b*3]; fp += st[b*3+1]; tt += st[b*3+2]; }
    out[0] = tp / (tp + fp + 0.001f);
    out[1] = tp / (tt + 0.001f);
    out[2] = tp / (tt + fp + 0.001f);
}

extern "C" void kernel_launch(void* const* d_in, const int* in_sizes, int n_in,
                              void* d_out, int out_size, void* d_ws, size_t ws_size,
                              hipStream_t stream) {
    const float* pred   = (const float*)d_in[0];
    const float* tgt    = (const float*)d_in[1];
    const float* logits = (const float*)d_in[2];
    const int*   tcls   = (const int*)d_in[3];
    int*   ws  = (int*)d_ws;
    float* out = (float*)d_out;

    zero_kernel<<<(WS_ZERO_N + 255) / 256, 256, 0, stream>>>(ws, WS_ZERO_N);

    dim3 grid(NHC, NPBLK, BS);
    iou_wmma_kernel<<<grid, 128, 0, stream>>>(pred, tgt, ws);

    score_kernel<<<BS * P_CH, 32, 0, stream>>>(logits, ws);
    match_kernel<<<BS, 32, 0, stream>>>(tcls, ws);
    final_kernel<<<1, 1, 0, stream>>>(ws, out);
}
